// BatchedNeRFMLP_49598282334732
// MI455X (gfx1250) — matrix operand order repaired and checked by reading, not compiled, not run on hardware
//
#include <hip/hip_runtime.h>

typedef _Float16 v16h __attribute__((ext_vector_type(16)));
typedef _Float16 v8h  __attribute__((ext_vector_type(8)));
typedef float    v8f  __attribute__((ext_vector_type(8)));

#define POS_IN   63
#define HID      128
#define TOTAL    8789
#define NPTS     65536
#define NBATCH   32

// param offsets (flat, per batch)
#define PW_OFF   0
#define PB_OFF   8064
#define SW_OFF   8192
#define SB_OFF   8320
#define CW_OFF   8321
#define CB_OFF   8786

#define WAVES       4
#define BLOCK       (WAVES * 32)                 // 128 threads
#define CHUNKS      128                          // workgroups per batch
#define PTS_PER_WG  (NPTS / CHUNKS)              // 512
#define ITERS       (PTS_PER_WG / (WAVES * 16))  // 8

// LDS row strides (halves), padded for bank-conflict-free row access, 16B-aligned
#define PW_RS  72      // s_pw rows (hid), cols 0..63 used (col 63 = pb bias)
#define PE_RS  72      // s_penc rows (pt), cols 0..63 used (col 63 = 1.0)
#define H_RS   168     // s_h rows (pt), cols 0..127 = h, 128..154 = dir_enc, 155 = 1.0, 156..159 = 0
#define W2_RS  168     // s_w2 rows (out 0..15), cols 0..159 used

__global__ __launch_bounds__(BLOCK) void nerf_mlp_kernel(
    const float* __restrict__ params,
    const float* __restrict__ points,
    const float* __restrict__ dirs,
    float* __restrict__ out)
{
    __shared__ _Float16 s_pw[HID * PW_RS];          // layer-1 A (pw + bias col), 18 KB
    __shared__ _Float16 s_penc[WAVES][16 * PE_RS];  // layer-1 B (pos_enc + 1), 9 KB
    __shared__ _Float16 s_h[WAVES][16 * H_RS];      // layer-2 B ([h ; dir_enc ; 1]), 21 KB
    __shared__ _Float16 s_w2[16 * W2_RS];           // layer-2 A (color/sigma weights + bias col), 5.25 KB

    const int tid = threadIdx.x;
    const int b   = blockIdx.y;
    const float* __restrict__ P = params + (size_t)b * TOTAL;

    // ---------------- stage per-batch params into LDS (f16) ----------------
    for (int idx = tid; idx < HID * 64; idx += BLOCK) {
        const int n = idx >> 6;          // hidden unit
        const int k = idx & 63;          // input feature (col 63 = bias)
        const float v = (k < POS_IN) ? P[PW_OFF + n * POS_IN + k] : P[PB_OFF + n];
        s_pw[n * PW_RS + k] = (_Float16)v;
    }
    for (int idx = tid; idx < 16 * 160; idx += BLOCK) {
        const int o = idx / 160;         // output row: 0..2 color, 3 sigma, 4..15 zero
        const int k = idx % 160;         // k: 0..127 h, 128..154 dir_enc, 155 bias, 156..159 pad
        float v = 0.0f;
        if (o < 3) {
            if (k < 155)       v = P[CW_OFF + o * 155 + k];
            else if (k == 155) v = P[CB_OFF + o];
        } else if (o == 3) {
            if (k < HID)       v = P[SW_OFF + k];
            else if (k == 155) v = P[SB_OFF];
        }
        s_w2[o * W2_RS + k] = (_Float16)v;
    }
    __syncthreads();

    const int lane  = tid & 31;
    const int wave  = tid >> 5;
    const int nl    = lane & 15;          // column index (point) this lane owns
    const int half  = lane >> 4;
    const int koffA = half << 3;          // A-layout K interleave base (0 or 8)
    const int koffB = half << 4;          // unused; B loads use koffA (same k order as A)

    // ---------------- hoist layer-1 A fragments (pw): 8 tiles x 2 k-steps ----------------
    v16h Af0[8], Af1[8];
#pragma unroll
    for (int t = 0; t < 8; ++t) {
        const _Float16* wr = &s_pw[(t * 16 + nl) * PW_RS];
        const v8h a0 = *(const v8h*)(wr + koffA);
        const v8h a1 = *(const v8h*)(wr + koffA + 16);
        const v8h a2 = *(const v8h*)(wr + 32 + koffA);
        const v8h a3 = *(const v8h*)(wr + 32 + koffA + 16);
        Af0[t] = __builtin_shufflevector(a0, a1, 0,1,2,3,4,5,6,7,8,9,10,11,12,13,14,15);
        Af1[t] = __builtin_shufflevector(a2, a3, 0,1,2,3,4,5,6,7,8,9,10,11,12,13,14,15);
    }
    // ---------------- hoist layer-2 A fragments (W2): 5 k-steps ----------------
    v16h Wf[5];
#pragma unroll
    for (int s = 0; s < 5; ++s) {
        const _Float16* wr = &s_w2[nl * W2_RS + s * 32];
        const v8h a0 = *(const v8h*)(wr + koffA);
        const v8h a1 = *(const v8h*)(wr + koffA + 16);
        Wf[s] = __builtin_shufflevector(a0, a1, 0,1,2,3,4,5,6,7,8,9,10,11,12,13,14,15);
    }

    for (int it = 0; it < ITERS; ++it) {
        const int p0 = blockIdx.x * PTS_PER_WG + it * (WAVES * 16) + wave * 16;

        __builtin_amdgcn_wave_barrier();   // order vs previous iteration's LDS reads

        // ---------------- encode pos_enc -> s_penc, dir_enc -> s_h cols 128..159 ----------
        {
            const int pt_i = lane >> 1;
            const int fh   = lane & 1;     // feature half
            const int p    = p0 + pt_i;
            const float* __restrict__ pt = points + ((size_t)b * NPTS + p) * 3;
            const float* __restrict__ dv = dirs   + ((size_t)b * NPTS + p) * 3;
            const float x0 = pt[0], x1 = pt[1], x2 = pt[2];
            const float d0 = dv[0], d1 = dv[1], d2 = dv[2];

            _Float16 e[32];   // pos_enc half
            _Float16 g[16];   // dir_enc half
            if (fh == 0) {
                // pos features 0..31: x0,x1,x2, sin_j j=0..28  (feature 3+j, j = c*10+f)
                e[0] = (_Float16)x0; e[1] = (_Float16)x1; e[2] = (_Float16)x2;
#pragma unroll
                for (int j = 0; j < 29; ++j) {
                    const int c = j / 10, f = j % 10;
                    const float xc = (c == 0) ? x0 : ((c == 1) ? x1 : x2);
                    e[3 + j] = (_Float16)__sinf(xc * (float)(1 << f));
                }
                // dir features 0..15: d0,d1,d2, sin_u u=0..11 (u = c*4+f), cos_0
                g[0] = (_Float16)d0; g[1] = (_Float16)d1; g[2] = (_Float16)d2;
#pragma unroll
                for (int u = 0; u < 12; ++u) {
                    const int c = u / 4, f = u % 4;
                    const float dc = (c == 0) ? d0 : ((c == 1) ? d1 : d2);
                    g[3 + u] = (_Float16)__sinf(dc * (float)(1 << f));
                }
                g[15] = (_Float16)__cosf(d0);                  // cos u=0
            } else {
                // pos features 32..63: sin_29, cos_j j=0..29, 1.0 (bias feature)
                e[0] = (_Float16)__sinf(x2 * 512.0f);          // j=29: c=2, f=9
#pragma unroll
                for (int j = 0; j < 30; ++j) {
                    const int c = j / 10, f = j % 10;
                    const float xc = (c == 0) ? x0 : ((c == 1) ? x1 : x2);
                    e[1 + j] = (_Float16)__cosf(xc * (float)(1 << f));
                }
                e[31] = (_Float16)1.0f;                        // feature 63 = 1 (pb bias)
                // dir features 16..31: cos_u u=1..11, 1.0 (cb/sb bias), zero pad
#pragma unroll
                for (int u = 1; u < 12; ++u) {
                    const int c = u / 4, f = u % 4;
                    const float dc = (c == 0) ? d0 : ((c == 1) ? d1 : d2);
                    g[u - 1] = (_Float16)__cosf(dc * (float)(1 << f));
                }
                g[11] = (_Float16)1.0f;                        // k=155 = 1 (bias)
                g[12] = (_Float16)0.0f; g[13] = (_Float16)0.0f;
                g[14] = (_Float16)0.0f; g[15] = (_Float16)0.0f;
            }

            _Float16* pe = &s_penc[wave][pt_i * PE_RS + fh * 32];
#pragma unroll
            for (int q = 0; q < 4; ++q) {
                v8h v;
#pragma unroll
                for (int i = 0; i < 8; ++i) v[i] = e[q * 8 + i];
                *(v8h*)(pe + q * 8) = v;
            }
            _Float16* de = &s_h[wave][pt_i * H_RS + 128 + fh * 16];
#pragma unroll
            for (int q = 0; q < 2; ++q) {
                v8h v;
#pragma unroll
                for (int i = 0; i < 8; ++i) v[i] = g[q * 8 + i];
                *(v8h*)(de + q * 8) = v;
            }
        }

        __builtin_amdgcn_wave_barrier();   // DS in-order per wave: stores precede loads below

        // ---------------- layer-1 B fragments (pos_enc, shared by all 8 tiles) ----------
        const _Float16* br = &s_penc[wave][nl * PE_RS];
        const v8h b00 = *(const v8h*)(br + koffA);
        const v8h b01 = *(const v8h*)(br + koffA + 16);
        const v8h b10 = *(const v8h*)(br + 32 + koffA);
        const v8h b11 = *(const v8h*)(br + 32 + koffA + 16);
        const v16h B10 = __builtin_shufflevector(b00, b01, 0,1,2,3,4,5,6,7,8,9,10,11,12,13,14,15);
        const v16h B11 = __builtin_shufflevector(b10, b11, 0,1,2,3,4,5,6,7,8,9,10,11,12,13,14,15);

        // ---------------- layer 1: h[hid][pt] tiles, relu in packed f16, store to s_h ----
        _Float16* hb = &s_h[wave][nl * H_RS + half * 8];
        const v8h zero8 = { (_Float16)0.0f, (_Float16)0.0f, (_Float16)0.0f, (_Float16)0.0f,
                            (_Float16)0.0f, (_Float16)0.0f, (_Float16)0.0f, (_Float16)0.0f };
#pragma unroll
        for (int t = 0; t < 8; ++t) {
            v8f acc = { 0.f, 0.f, 0.f, 0.f, 0.f, 0.f, 0.f, 0.f };   // SRC2 = inline 0
            acc = __builtin_amdgcn_wmma_f32_16x16x32_f16(false, Af0[t], false, B10,
                                                         (short)0, acc, false, false);
            acc = __builtin_amdgcn_wmma_f32_16x16x32_f16(false, Af1[t], false, B11,
                                                         (short)0, acc, false, false);
            // convert first (v_cvt_pk_f16_f32 pairs), then clamp packed (v_pk_max_num_f16)
            v8h hv;
#pragma unroll
            for (int r = 0; r < 8; ++r) hv[r] = (_Float16)acc[r];
            hv = __builtin_elementwise_max(hv, zero8);               // relu (bias via k=63)
            *(v8h*)(hb + t * 16) = hv;    // contiguous: rows hid = t*16 + half*8 .. +7 of point nl
        }

        __builtin_amdgcn_wave_barrier();   // h stores precede layer-2 B reads (DS in-order)

        // ---------------- layer 2: [c0,c1,c2,sigma][pt] = W2 x [h ; dir_enc ; 1] ----------
        v8f o4 = { 0.f, 0.f, 0.f, 0.f, 0.f, 0.f, 0.f, 0.f };
        const _Float16* hr = &s_h[wave][nl * H_RS];
#pragma unroll
        for (int s = 0; s < 5; ++s) {
            const v8h c0 = *(const v8h*)(hr + s * 32 + koffA);
            const v8h c1 = *(const v8h*)(hr + s * 32 + koffA + 16);
            const v16h B2 = __builtin_shufflevector(c0, c1, 0,1,2,3,4,5,6,7,8,9,10,11,12,13,14,15);
            o4 = __builtin_amdgcn_wmma_f32_16x16x32_f16(false, Wf[s], false, B2,
                                                        (short)0, o4, false, false);
        }

        // lanes 0..15 hold rows 0..7 of column pt=lane: o4[0..3] = c0,c1,c2,sigma
        if (lane < 16) {
            const int p = p0 + lane;
            const float C0 = 1.0f / (1.0f + __expf(-o4[0]));
            const float C1 = 1.0f / (1.0f + __expf(-o4[1]));
            const float C2 = 1.0f / (1.0f + __expf(-o4[2]));
            const float S  = o4[3];
            float4 o = make_float4(C0, C1, C2, S);
            *(float4*)(out + ((size_t)b * NPTS + p) * 4) = o;   // 16 lanes -> 256B coalesced
        }
    }
}

extern "C" void kernel_launch(void* const* d_in, const int* in_sizes, int n_in,
                              void* d_out, int out_size, void* d_ws, size_t ws_size,
                              hipStream_t stream) {
    (void)in_sizes; (void)n_in; (void)d_ws; (void)ws_size; (void)out_size;
    const float* params = (const float*)d_in[0];
    const float* points = (const float*)d_in[1];
    const float* dirs   = (const float*)d_in[2];
    float* out = (float*)d_out;

    dim3 grid(CHUNKS, NBATCH, 1);
    dim3 block(BLOCK, 1, 1);
    nerf_mlp_kernel<<<grid, block, 0, stream>>>(params, points, dirs, out);
}
